// DinoDecoderBlock_56100862821031
// MI455X (gfx1250) — matrix operand order repaired and checked by reading, not compiled, not run on hardware
//
#include <hip/hip_runtime.h>

// ---------------------------------------------------------------------------
// DinoDecoderBlock on MI455X (gfx1250): bf16 WMMA everywhere, flash attention,
// double-buffered async global->LDS staging (GLOBAL_LOAD_ASYNC_TO_LDS_B128,
// ASYNCcnt-pipelined so tile k+1 copies overlap tile k WMMAs).
// ---------------------------------------------------------------------------

#define BATCH 8
#define SEQ   1024
#define DIMC  768
#define NHEAD 12
#define HDIM  64
#define HIDC  3072
#define MTOT  (BATCH * SEQ)
#define ATT_SCALE 0.125f   // 64^-0.5

typedef __bf16 bf16_t;
typedef __attribute__((ext_vector_type(16))) __bf16 v16bf;
typedef __attribute__((ext_vector_type(8)))  __bf16 v8bf;
typedef __attribute__((ext_vector_type(8)))  float  v8f;

// float -> bf16 round-to-nearest-even (bit-exact, integer path only)
__device__ __forceinline__ bf16_t f2bf(float f) {
  union { float f; unsigned u; } c; c.f = f;
  unsigned u = c.u + 0x7FFFu + ((c.u >> 16) & 1u);
  unsigned short h = (unsigned short)(u >> 16);
  return __builtin_bit_cast(bf16_t, h);
}

__device__ __forceinline__ v8f vzero8() {
  v8f v;
#pragma unroll
  for (int i = 0; i < 8; ++i) v[i] = 0.0f;
  return v;
}

// ---- CDNA5 async global->LDS copy (bypasses VGPRs, tracked by ASYNCcnt) ----
// LDS destination: low 32 bits of the generic pointer == LDS byte offset
// (shared-aperture flat addresses truncate to LDS_ADDR per the ISA).
__device__ __forceinline__ void async_g2l_b128(void* lds_dst, const void* gsrc) {
  unsigned l = (unsigned)(uintptr_t)lds_dst;
  unsigned long long g = (unsigned long long)(uintptr_t)gsrc;
  asm volatile("global_load_async_to_lds_b128 %0, %1, off"
               :: "v"(l), "v"(g) : "memory");
}
template <int N>
__device__ __forceinline__ void wait_asynccnt() {
  asm volatile("s_wait_asynccnt %0" :: "n"(N) : "memory");
}

// A-matrix fragment (16x32 bf16) from an LDS row base (row = M, contiguous K).
// lanes 0-15: VGPR0-7 hold K=0..15
// lanes16-31: VGPR0-7 hold K=16..31
__device__ __forceinline__ v16bf load_afrag(const bf16_t* row, int lh) {
  v8bf lo = *(const v8bf*)(row + lh * 16);
  v8bf hi = *(const v8bf*)(row + lh * 16 + 8);
  v16bf a;
#pragma unroll
  for (int i = 0; i < 8; ++i) { a[i] = lo[i]; a[i + 8] = hi[i]; }
  return a;
}

// B-matrix fragment (32x16 bf16): lane = column n, K contiguous.
// lanes 0-15 hold K=0..15, lanes 16-31 hold K=16..31 (2 per VGPR).
__device__ __forceinline__ v16bf load_bfrag(const bf16_t* row, int lh) {
  const bf16_t* p = row + lh * 16;
  v8bf lo = *(const v8bf*)(p);
  v8bf hi = *(const v8bf*)(p + 8);
  v16bf b;
#pragma unroll
  for (int i = 0; i < 8; ++i) { b[i] = lo[i]; b[i + 8] = hi[i]; }
  return b;
}

// ---------------------------------------------------------------------------
// f32 -> bf16 conversion (weights)
// ---------------------------------------------------------------------------
__global__ void cvt_f32_bf16(const float* __restrict__ in,
                             bf16_t* __restrict__ out, int n) {
  int i = blockIdx.x * blockDim.x + threadIdx.x;
  if (i < n) out[i] = f2bf(in[i]);
}

// ---------------------------------------------------------------------------
// LayerNorm (row = 768) producing bf16
// ---------------------------------------------------------------------------
__global__ __launch_bounds__(256) void ln_bf16(const float* __restrict__ x,
                                               const float* __restrict__ g,
                                               const float* __restrict__ bt,
                                               bf16_t* __restrict__ out) {
  const int row = blockIdx.x;
  const int tid = threadIdx.x;
  const float* xr = x + (size_t)row * DIMC;
  float v0 = xr[tid], v1 = xr[tid + 256], v2 = xr[tid + 512];
  float s  = v0 + v1 + v2;
  float ss = v0 * v0 + v1 * v1 + v2 * v2;
  __shared__ float rs[8], rss[8];
#pragma unroll
  for (int off = 16; off; off >>= 1) {
    s  += __shfl_xor(s, off, 32);
    ss += __shfl_xor(ss, off, 32);
  }
  if ((tid & 31) == 0) { rs[tid >> 5] = s; rss[tid >> 5] = ss; }
  __syncthreads();
  float ts = 0.f, tss = 0.f;
#pragma unroll
  for (int i = 0; i < 8; ++i) { ts += rs[i]; tss += rss[i]; }
  const float mu   = ts * (1.0f / DIMC);
  const float var  = tss * (1.0f / DIMC) - mu * mu;
  const float rstd = rsqrtf(var + 1e-5f);
  bf16_t* orow = out + (size_t)row * DIMC;
  orow[tid]       = f2bf((v0 - mu) * rstd * g[tid]       + bt[tid]);
  orow[tid + 256] = f2bf((v1 - mu) * rstd * g[tid + 256] + bt[tid + 256]);
  orow[tid + 512] = f2bf((v2 - mu) * rstd * g[tid + 512] + bt[tid + 512]);
}

// ---------------------------------------------------------------------------
// WMMA GEMM: out[m,n] = sum_k A[m,k] * W[n,k] + bias[n]  (+ epilogue)
// block tile 64x64, 4 waves; wave w owns rows w*16..w*16+15 x all 64 cols
// (4 accumulators, one shared A-fragment per K-stage -> 4 WMMA per stage).
// Tiles double-buffered via async global->LDS copies: stage i+1 copies are in
// flight (ASYNCcnt) while stage i is multiplied; s_wait_asynccnt 4 retires
// the previous stage (4 async b128 per thread per stage, in-order completion).
// ---------------------------------------------------------------------------
enum { EP_QKV = 0, EP_HEADS = 1, EP_RESID = 2, EP_GELU = 3 };

template <int EP>
__global__ __launch_bounds__(128) void gemm_bf16(
    const bf16_t* __restrict__ A, const bf16_t* __restrict__ W,
    const float* __restrict__ bias, const float* __restrict__ resid,
    bf16_t* __restrict__ ob0, bf16_t* __restrict__ ob1,
    bf16_t* __restrict__ ob2, float* __restrict__ of,
    int K, int Nout) {
  __shared__ bf16_t As[2][64][40];   // 80B row pitch: 16B aligned, bank-spread
  __shared__ bf16_t Bs[2][64][40];

  const int tid  = threadIdx.x;
  const int wave = tid >> 5;
  const int lane = tid & 31;
  const int lh   = lane >> 4;     // lane half
  const int l15  = lane & 15;
  const int m0   = blockIdx.y * 64;
  const int n0   = blockIdx.x * 64;
  const int wm   = wave * 16;

  const int lr = tid >> 1;        // 0..63 staging row
  const int lc = (tid & 1) * 16;  // 0 or 16

  const bf16_t* Arow = A + (size_t)(m0 + lr) * K + lc;
  const bf16_t* Wrow = W + (size_t)(n0 + lr) * K + lc;

  v8f c[4];
#pragma unroll
  for (int t = 0; t < 4; ++t) c[t] = vzero8();

  const int nk = K >> 5;

  // prologue: stage 0 into buffer 0
  async_g2l_b128(&As[0][lr][lc],     Arow);
  async_g2l_b128(&As[0][lr][lc + 8], Arow + 8);
  async_g2l_b128(&Bs[0][lr][lc],     Wrow);
  async_g2l_b128(&Bs[0][lr][lc + 8], Wrow + 8);

  for (int i = 0; i < nk; ++i) {
    const int cur = i & 1;
    if (i + 1 < nk) {  // issue stage i+1 into the other buffer, keep flying
      const bf16_t* an = Arow + (size_t)(i + 1) * 32;
      const bf16_t* wn = Wrow + (size_t)(i + 1) * 32;
      async_g2l_b128(&As[cur ^ 1][lr][lc],     an);
      async_g2l_b128(&As[cur ^ 1][lr][lc + 8], an + 8);
      async_g2l_b128(&Bs[cur ^ 1][lr][lc],     wn);
      async_g2l_b128(&Bs[cur ^ 1][lr][lc + 8], wn + 8);
      if (i + 2 < nk) __builtin_prefetch(Arow + (size_t)(i + 2) * 32, 0, 1);
      wait_asynccnt<4>();  // stage i landed (in-order), stage i+1 still flying
    } else {
      wait_asynccnt<0>();
    }
    __syncthreads();       // everyone's stage-i copies visible

    v16bf a = load_afrag(&As[cur][wm + l15][0], lh);
#pragma unroll
    for (int t = 0; t < 4; ++t) {
      v16bf bb = load_bfrag(&Bs[cur][t * 16 + l15][0], lh);
      c[t] = __builtin_amdgcn_wmma_f32_16x16x32_bf16(false, a, false, bb,
                                                     (short)0, c[t], false,
                                                     false);
    }
    __syncthreads();       // readers done before next issue overwrites buffer
  }

  // epilogue (C layout: VGPR j -> row j (+8 for upper half), lane = column)
#pragma unroll
  for (int t = 0; t < 4; ++t) {
    const int nn = n0 + t * 16 + l15;
    const float bv = bias[nn];
#pragma unroll
    for (int j = 0; j < 8; ++j) {
      const int mm = m0 + wm + lh * 8 + j;
      float v = c[t][j] + bv;
      if (EP == EP_RESID) {
        of[(size_t)mm * Nout + nn] = resid[(size_t)mm * Nout + nn] + v;
      } else if (EP == EP_GELU) {
        float gg = 0.5f * v * (1.0f + erff(v * 0.70710678118654752f));
        ob0[(size_t)mm * Nout + nn] = f2bf(gg);
      } else if (EP == EP_HEADS) {
        const int hh = nn >> 6, dd = nn & 63;
        const int bb = mm >> 10, tok = mm & (SEQ - 1);
        ob0[(((size_t)bb * NHEAD + hh) * SEQ + tok) * HDIM + dd] = f2bf(v);
      } else {  // EP_QKV: split 3C columns into Q/K/V head-major buffers
        const int sel = nn / DIMC, cc = nn % DIMC;
        const int hh = cc >> 6, dd = cc & 63;
        const int bb = mm >> 10, tok = mm & (SEQ - 1);
        bf16_t* dst = (sel == 0) ? ob0 : ((sel == 1) ? ob1 : ob2);
        dst[(((size_t)bb * NHEAD + hh) * SEQ + tok) * HDIM + dd] = f2bf(v);
      }
    }
  }
}

// ---------------------------------------------------------------------------
// Flash attention: 1 block = (b, h, 64 q rows); 4 waves x 16 q rows.
// Online softmax over 64-key tiles; S and P*V via WMMA.
// Q/K tiles staged asynchronously; V is transposed through VGPRs into LDS
// (the transpose VALU work overlaps the in-flight async K copies).
// MASKED adds similarities[q,k] and -inf mask (sparse cross attention).
// ---------------------------------------------------------------------------
template <bool MASKED>
__global__ __launch_bounds__(128) void attn_fused(
    const bf16_t* __restrict__ Qg, const bf16_t* __restrict__ Kg,
    const bf16_t* __restrict__ Vg, const float* __restrict__ sim,
    const unsigned char* __restrict__ mask, bf16_t* __restrict__ Om) {
  __shared__ bf16_t Qs[64][72];       // [q ][d]  144B pitch, 16B aligned
  __shared__ bf16_t Ks[64][72];       // [k ][d]
  __shared__ bf16_t Vt[64][72];       // [d ][k]  (V transposed)
  __shared__ bf16_t Ps[4][16][72];    // per-wave P staging (C->A relayout)

  const int b = blockIdx.z, h = blockIdx.y, qt = blockIdx.x;
  const int tid  = threadIdx.x;
  const int wave = tid >> 5;
  const int lane = tid & 31;
  const int lh   = lane >> 4;
  const int l15  = lane & 15;

  const size_t headBase = ((size_t)(b * NHEAD + h)) * SEQ * HDIM;
  const int r  = tid >> 1;            // 0..63 staging row
  const int c0 = (tid & 1) * 32;

  {  // resident Q tile (async; completion covered by first loop wait+barrier)
    const bf16_t* src = Qg + headBase + (size_t)qt * 64 * HDIM;
#pragma unroll
    for (int c = 0; c < 32; c += 8)
      async_g2l_b128(&Qs[r][c0 + c], &src[r * HDIM + c0 + c]);
  }

  v8f o[4];
#pragma unroll
  for (int t = 0; t < 4; ++t) o[t] = vzero8();
  float rm[8], rl[8];
#pragma unroll
  for (int j = 0; j < 8; ++j) { rm[j] = -INFINITY; rl[j] = 0.f; }

  const int q0 = qt * 64 + wave * 16;

  for (int kt = 0; kt < SEQ / 64; ++kt) {
    __syncthreads();  // protect Ks/Vt against previous-iteration readers
    {
      const bf16_t* ks = Kg + headBase + (size_t)kt * 64 * HDIM;
      const bf16_t* vs = Vg + headBase + (size_t)kt * 64 * HDIM;
#pragma unroll
      for (int c = 0; c < 32; c += 8)
        async_g2l_b128(&Ks[r][c0 + c], &ks[r * HDIM + c0 + c]);
#pragma unroll
      for (int c = 0; c < 32; c += 8) {
        v8bf vv = *(const v8bf*)&vs[r * HDIM + c0 + c];
#pragma unroll
        for (int e = 0; e < 8; ++e) Vt[c0 + c + e][r] = vv[e];
      }
    }
    wait_asynccnt<0>();   // own async copies (Q on first iter, K always) done
    __syncthreads();

    // S = Q K^T : 16x64 per wave (4 n-tiles, 2 k-steps over d)
    v8f s[4];
#pragma unroll
    for (int t = 0; t < 4; ++t) s[t] = vzero8();
#pragma unroll
    for (int kk = 0; kk < 2; ++kk) {
      v16bf a = load_afrag(&Qs[wave * 16 + l15][kk * 32], lh);
#pragma unroll
      for (int t = 0; t < 4; ++t) {
        v16bf bb = load_bfrag(&Ks[t * 16 + l15][kk * 32], lh);
        s[t] = __builtin_amdgcn_wmma_f32_16x16x32_bf16(false, a, false, bb,
                                                       (short)0, s[t], false,
                                                       false);
      }
    }

    // scale + (sim, mask) + tile max
    float pv[4][8], cm[8];
#pragma unroll
    for (int j = 0; j < 8; ++j) cm[j] = -INFINITY;
    const int qb = q0 + lh * 8;
#pragma unroll
    for (int t = 0; t < 4; ++t) {
      const int key = kt * 64 + t * 16 + l15;
#pragma unroll
      for (int j = 0; j < 8; ++j) {
        float v = s[t][j] * ATT_SCALE;
        if (MASKED) {
          const size_t si = (size_t)(qb + j) * SEQ + key;
          v += sim[si];
          if (!mask[si]) v = -INFINITY;
        }
        pv[t][j] = v;
        cm[j] = fmaxf(cm[j], v);
      }
    }
#pragma unroll
    for (int off = 1; off < 16; off <<= 1)
#pragma unroll
      for (int j = 0; j < 8; ++j)
        cm[j] = fmaxf(cm[j], __shfl_xor(cm[j], off, 32));

    float corr[8], psum[8];
#pragma unroll
    for (int j = 0; j < 8; ++j) {
      const float nm = fmaxf(rm[j], cm[j]);
      corr[j] = __expf(rm[j] - nm);
      rm[j] = nm;
      psum[j] = 0.f;
    }
#pragma unroll
    for (int t = 0; t < 4; ++t)
#pragma unroll
      for (int j = 0; j < 8; ++j) {
        const float p = __expf(pv[t][j] - rm[j]);
        psum[j] += p;
        Ps[wave][j + lh * 8][t * 16 + l15] = f2bf(p);  // C -> A relayout
      }
#pragma unroll
    for (int off = 1; off < 16; off <<= 1)
#pragma unroll
      for (int j = 0; j < 8; ++j) psum[j] += __shfl_xor(psum[j], off, 32);
#pragma unroll
    for (int j = 0; j < 8; ++j) rl[j] = rl[j] * corr[j] + psum[j];
#pragma unroll
    for (int t = 0; t < 4; ++t)
#pragma unroll
      for (int j = 0; j < 8; ++j) o[t][j] *= corr[j];

    // O += P V  (k over 64 keys = 2 steps)
#pragma unroll
    for (int kk = 0; kk < 2; ++kk) {
      v16bf a = load_afrag(&Ps[wave][l15][kk * 32], lh);
#pragma unroll
      for (int t = 0; t < 4; ++t) {
        v16bf bb = load_bfrag(&Vt[t * 16 + l15][kk * 32], lh);
        o[t] = __builtin_amdgcn_wmma_f32_16x16x32_bf16(false, a, false, bb,
                                                       (short)0, o[t], false,
                                                       false);
      }
    }
  }

  // normalize + write merged layout [B,N,C] (head-major columns)
  const size_t ob = (size_t)b * SEQ * DIMC + (size_t)h * HDIM;
#pragma unroll
  for (int t = 0; t < 4; ++t) {
    const int d = t * 16 + l15;
#pragma unroll
    for (int j = 0; j < 8; ++j) {
      const int q = q0 + lh * 8 + j;
      Om[ob + (size_t)q * DIMC + d] = f2bf(o[t][j] / rl[j]);
    }
  }
}

// ---------------------------------------------------------------------------
// host orchestration
// ---------------------------------------------------------------------------
extern "C" void kernel_launch(void* const* d_in, const int* in_sizes, int n_in,
                              void* d_out, int out_size, void* d_ws,
                              size_t ws_size, hipStream_t stream) {
  (void)in_sizes; (void)n_in; (void)out_size; (void)ws_size;

  const float* x   = (const float*)d_in[0];
  const float* y   = (const float*)d_in[1];
  const unsigned char* mask = (const unsigned char*)d_in[4];
  const float* sim = (const float*)d_in[5];
  const float* norm1_w = (const float*)d_in[6];
  const float* norm1_b = (const float*)d_in[7];
  const float* norm2_w = (const float*)d_in[8];
  const float* norm2_b = (const float*)d_in[9];
  const float* norm3_w = (const float*)d_in[10];
  const float* norm3_b = (const float*)d_in[11];
  const float* normy_w = (const float*)d_in[12];
  const float* normy_b = (const float*)d_in[13];
  const float* qkv_w   = (const float*)d_in[14];
  const float* qkv_b   = (const float*)d_in[15];
  const float* aproj_w = (const float*)d_in[16];
  const float* aproj_b = (const float*)d_in[17];
  const float* pq_w    = (const float*)d_in[18];
  const float* pq_b    = (const float*)d_in[19];
  const float* pk_w    = (const float*)d_in[20];
  const float* pk_b    = (const float*)d_in[21];
  const float* pv_w    = (const float*)d_in[22];
  const float* pv_b    = (const float*)d_in[23];
  const float* cproj_w = (const float*)d_in[24];
  const float* cproj_b = (const float*)d_in[25];
  const float* fc1_w   = (const float*)d_in[26];
  const float* fc1_b   = (const float*)d_in[27];
  const float* fc2_w   = (const float*)d_in[28];
  const float* fc2_b   = (const float*)d_in[29];

  float* xout = (float*)d_out;  // running residual stream, final output

  char* ws = (char*)d_ws;
  size_t off = 0;
  auto alloc = [&](size_t elems) -> bf16_t* {
    bf16_t* p = (bf16_t*)(ws + off);
    off = (off + elems * sizeof(bf16_t) + 255) & ~(size_t)255;
    return p;
  };
  bf16_t* w_qkv = alloc((size_t)3 * DIMC * DIMC);
  bf16_t* w_ap  = alloc((size_t)DIMC * DIMC);
  bf16_t* w_pq  = alloc((size_t)DIMC * DIMC);
  bf16_t* w_pk  = alloc((size_t)DIMC * DIMC);
  bf16_t* w_pv  = alloc((size_t)DIMC * DIMC);
  bf16_t* w_cp  = alloc((size_t)DIMC * DIMC);
  bf16_t* w_fc1 = alloc((size_t)HIDC * DIMC);
  bf16_t* w_fc2 = alloc((size_t)DIMC * HIDC);
  bf16_t* a_bf  = alloc((size_t)MTOT * DIMC);  // LN output / GEMM A
  bf16_t* yn_bf = alloc((size_t)MTOT * DIMC);  // LN(y)
  bf16_t* Qb    = alloc((size_t)MTOT * DIMC);  // [B,H,N,D]
  bf16_t* Kb    = alloc((size_t)MTOT * DIMC);
  bf16_t* Vb    = alloc((size_t)MTOT * DIMC);
  bf16_t* ao    = alloc((size_t)MTOT * DIMC);  // merged attn out [B,N,C]
  bf16_t* hg    = alloc((size_t)MTOT * HIDC);  // gelu(fc1)

  auto cvt = [&](const float* s, bf16_t* d, int n) {
    cvt_f32_bf16<<<(n + 255) / 256, 256, 0, stream>>>(s, d, n);
  };
  cvt(qkv_w,   w_qkv, 3 * DIMC * DIMC);
  cvt(aproj_w, w_ap,  DIMC * DIMC);
  cvt(pq_w,    w_pq,  DIMC * DIMC);
  cvt(pk_w,    w_pk,  DIMC * DIMC);
  cvt(pv_w,    w_pv,  DIMC * DIMC);
  cvt(cproj_w, w_cp,  DIMC * DIMC);
  cvt(fc1_w,   w_fc1, HIDC * DIMC);
  cvt(fc2_w,   w_fc2, DIMC * HIDC);

  const dim3 gC(DIMC / 64, MTOT / 64);
  const dim3 gA(SEQ / 64, NHEAD, BATCH);

  // --- self attention ---
  ln_bf16<<<MTOT, 256, 0, stream>>>(x, norm1_w, norm1_b, a_bf);
  gemm_bf16<EP_QKV><<<dim3((3 * DIMC) / 64, MTOT / 64), 128, 0, stream>>>(
      a_bf, w_qkv, qkv_b, nullptr, Qb, Kb, Vb, nullptr, DIMC, 3 * DIMC);
  attn_fused<false><<<gA, 128, 0, stream>>>(Qb, Kb, Vb, nullptr, nullptr, ao);
  gemm_bf16<EP_RESID><<<gC, 128, 0, stream>>>(
      ao, w_ap, aproj_b, x, nullptr, nullptr, nullptr, xout, DIMC, DIMC);

  // --- sparse cross attention ---
  ln_bf16<<<MTOT, 256, 0, stream>>>(y, normy_w, normy_b, yn_bf);
  ln_bf16<<<MTOT, 256, 0, stream>>>(xout, norm2_w, norm2_b, a_bf);
  gemm_bf16<EP_HEADS><<<gC, 128, 0, stream>>>(
      a_bf, w_pq, pq_b, nullptr, Qb, nullptr, nullptr, nullptr, DIMC, DIMC);
  gemm_bf16<EP_HEADS><<<gC, 128, 0, stream>>>(
      yn_bf, w_pk, pk_b, nullptr, Kb, nullptr, nullptr, nullptr, DIMC, DIMC);
  gemm_bf16<EP_HEADS><<<gC, 128, 0, stream>>>(
      yn_bf, w_pv, pv_b, nullptr, Vb, nullptr, nullptr, nullptr, DIMC, DIMC);
  attn_fused<true><<<gA, 128, 0, stream>>>(Qb, Kb, Vb, sim, mask, ao);
  gemm_bf16<EP_RESID><<<gC, 128, 0, stream>>>(
      ao, w_cp, cproj_b, xout, nullptr, nullptr, nullptr, xout, DIMC, DIMC);

  // --- mlp ---
  ln_bf16<<<MTOT, 256, 0, stream>>>(xout, norm3_w, norm3_b, a_bf);
  gemm_bf16<EP_GELU><<<dim3(HIDC / 64, MTOT / 64), 128, 0, stream>>>(
      a_bf, w_fc1, fc1_b, nullptr, hg, nullptr, nullptr, nullptr, DIMC, HIDC);
  gemm_bf16<EP_RESID><<<gC, 128, 0, stream>>>(
      hg, w_fc2, fc2_b, xout, nullptr, nullptr, nullptr, xout, HIDC, DIMC);
}
